// SetTransform_893353198455
// MI455X (gfx1250) — compile-verified, matrix operand order checked by reading, not run on hardware
//
#include <hip/hip_runtime.h>
#include <hip/hip_bf16.h>

typedef __attribute__((ext_vector_type(16))) __bf16 bf16x16;
typedef __attribute__((ext_vector_type(8)))  __bf16 bf16x8;
typedef __attribute__((ext_vector_type(8)))  float  f32x8;

#define BN_EPS 1e-5f
#define LDST 40   // padded LDS row stride (elements): 80B rows -> conflict-free b128 gathers

// ---------- helpers ----------
// Exact float atomic max via sign-split integer atomics (order-independent, deterministic).
__device__ inline void atomicMaxF(float* addr, float val) {
  if (val >= 0.0f) atomicMax((int*)addr, __float_as_int(val));
  else             atomicMin((unsigned int*)addr, __float_as_uint(val));
}

// Async copy of 16 bytes (one lane) from global memory directly into LDS.
// GV mode: 64-bit address in a VGPR pair, SADDR=off; VDST = LDS byte offset.
// Tracked by ASYNCcnt (gfx1250 async path) -- no VGPR staging.
__device__ inline void async_copy16(unsigned lds_off, const void* g) {
  asm volatile("global_load_async_to_lds_b128 %0, %1, off"
               :: "v"(lds_off), "v"((unsigned long long)(uintptr_t)g)
               : "memory");
}
__device__ inline void wait_async_le4() {
  asm volatile("s_wait_asynccnt 0x4" ::: "memory");
}
__device__ inline void wait_async_0() {
  asm volatile("s_wait_asynccnt 0x0" ::: "memory");
}

// Gather a 16x32 bf16 WMMA operand fragment from an LDS slab stored [row][k0..31].
// ISA 16-bit A-layout: lanes 0-15 hold rows, K 0-7 & 16-23; lanes 16-31 hold K 8-15 & 24-31.
// B operand (Bt slab, [outcol][k]) uses the identical gather with row := output column.
__device__ inline bf16x16 load_frag(const __bf16* lds, int base_row, int lane) {
  const int r  = lane & 15;
  const int hi = lane >> 4;
  const __bf16* p = lds + (base_row + r) * LDST + hi * 8;
  bf16x8 lo = *(const bf16x8*)(p);        // K = hi*8 + 0..7
  bf16x8 h8 = *(const bf16x8*)(p + 16);   // K = 16 + hi*8 + 0..7
  bf16x16 f;
#pragma unroll
  for (int i = 0; i < 8; ++i) { f[i] = lo[i]; f[i + 8] = h8[i]; }
  return f;
}

// ---------- GEMM: C[M,N] = A[M,K](bf16) @ Bt[N,K](bf16)^T + cp[N], epilogue ReLU-store or set-max ----
// 256 threads / 8 waves; 128x128 tile; K-step 32; double-buffered LDS fed by async loads.
template <bool RELU_STORE>
__global__ __launch_bounds__(256)
void gemm_bf16(const __bf16* __restrict__ A, const __bf16* __restrict__ Bt,
               const float* __restrict__ cp, __bf16* __restrict__ out,
               float* __restrict__ outmax, int K, int N) {
  __shared__ __bf16 Alds[2][128 * LDST];
  __shared__ __bf16 Blds[2][128 * LDST];
  const int tid  = threadIdx.x;
  const int lane = tid & 31;
  const int wave = tid >> 5;
  const int wm = wave & 1;                 // 64-row group within tile
  const int wn = wave >> 1;                // 32-col group within tile
  const int row_base = blockIdx.y * 128;
  const int col_base = blockIdx.x * 128;

  f32x8 acc[4][2];
#pragma unroll
  for (int i = 0; i < 4; ++i)
#pragma unroll
    for (int j = 0; j < 2; ++j) acc[i][j] = (f32x8)(0.0f);

  const int lr  = tid >> 1;   // 0..127: slab row (A) / output column (Bt)
  const int seg = tid & 1;    // 16-element K chunk

  // Per-lane LDS byte offsets (low 32 bits of the generic LDS address = offset in allocation).
  const unsigned aoff[2] = {
      (unsigned)(uintptr_t)&Alds[0][lr * LDST + seg * 16],
      (unsigned)(uintptr_t)&Alds[1][lr * LDST + seg * 16]};
  const unsigned boff[2] = {
      (unsigned)(uintptr_t)&Blds[0][lr * LDST + seg * 16],
      (unsigned)(uintptr_t)&Blds[1][lr * LDST + seg * 16]};
  const __bf16* gA = A  + (size_t)(row_base + lr) * K + seg * 16;
  const __bf16* gB = Bt + (size_t)(col_base + lr) * K + seg * 16;

  const int niter = K >> 5;
  // issue stage 'it' into buffer it&1: 4 async b128 copies per lane (32B of A + 32B of B)
  auto issue = [&](int it) {
    const int buf = it & 1;
    const __bf16* ga = gA + it * 32;
    const __bf16* gb = gB + it * 32;
    async_copy16(aoff[buf],      ga);
    async_copy16(aoff[buf] + 16, ga + 8);
    async_copy16(boff[buf],      gb);
    async_copy16(boff[buf] + 16, gb + 8);
  };

  issue(0);
  for (int it = 0; it < niter; ++it) {
    const int buf = it & 1;
    if (it + 1 < niter) {
      issue(it + 1);        // prefetch next stage into the other buffer
      wait_async_le4();     // in-order completion: stage 'it' (4 older ops) has landed
    } else {
      wait_async_0();
    }
    __syncthreads();        // all waves' async writes to buf are visible

    bf16x16 af[4];
#pragma unroll
    for (int i = 0; i < 4; ++i) af[i] = load_frag(Alds[buf], wm * 64 + i * 16, lane);
    bf16x16 bfr[2];
#pragma unroll
    for (int j = 0; j < 2; ++j) bfr[j] = load_frag(Blds[buf], wn * 32 + j * 16, lane);
#pragma unroll
    for (int i = 0; i < 4; ++i)
#pragma unroll
      for (int j = 0; j < 2; ++j)
        acc[i][j] = __builtin_amdgcn_wmma_f32_16x16x32_bf16(
            false, af[i], false, bfr[j], (short)0, acc[i][j], false, false);
    __syncthreads();        // done reading buf before it is overwritten next+1 stage
  }

  // C/D layout: lane = (m>=8)*16 + n, vgpr v = m%8 within each 16x16 tile.
  const int lc  = lane & 15;
  const int lhi = lane >> 4;
  if (RELU_STORE) {
#pragma unroll
    for (int j = 0; j < 2; ++j) {
      const int n = col_base + wn * 32 + j * 16 + lc;
      const float bias = cp[n];
#pragma unroll
      for (int i = 0; i < 4; ++i) {
        const int m = row_base + wm * 64 + i * 16 + lhi * 8;
#pragma unroll
        for (int v = 0; v < 8; ++v) {
          float y = acc[i][j][v] + bias;
          y = y > 0.0f ? y : 0.0f;
          out[(size_t)(m + v) * N + n] = (__bf16)y;
        }
      }
    }
  } else {
    // Fused max over the set axis: 2048 rows/batch, 128 | 2048 so the whole tile is one batch.
    const int b = row_base >> 11;
#pragma unroll
    for (int j = 0; j < 2; ++j) {
      const int n = col_base + wn * 32 + j * 16 + lc;
      float mx = -3.4e38f;
#pragma unroll
      for (int i = 0; i < 4; ++i)
#pragma unroll
        for (int v = 0; v < 8; ++v) mx = fmaxf(mx, acc[i][j][v]);
      mx = fmaxf(mx, __shfl_xor(mx, 16, 32));   // combine lane halves (same column)
      if (lhi == 0) atomicMaxF(outmax + (size_t)b * N + n, mx + cp[n]);
    }
  }
}

// ---------- column stats pass 1: per-(row-chunk) partial sum / sumsq, fixed order ----------
__global__ __launch_bounds__(256)
void colstats_part(const __bf16* __restrict__ X, float* __restrict__ psum,
                   float* __restrict__ psumsq, int D, int rows_per_blk) {
  const int tid = threadIdx.x;
  const int col = blockIdx.x * 64 + (tid & 63);
  const int rl  = tid >> 6;   // 0..3
  const int r0  = blockIdx.y * rows_per_blk;
  float s = 0.f, s2 = 0.f;
  for (int r = r0 + rl; r < r0 + rows_per_blk; r += 4) {
    float v = (float)X[(size_t)r * D + col];
    s += v; s2 += v * v;
  }
  __shared__ float ls[256], ls2[256];
  ls[tid] = s; ls2[tid] = s2;
  __syncthreads();
  if (rl == 0) {
    for (int t = 1; t < 4; ++t) { s += ls[tid + 64 * t]; s2 += ls2[tid + 64 * t]; }
    psum[(size_t)blockIdx.y * D + col]   = s;
    psumsq[(size_t)blockIdx.y * D + col] = s2;
  }
}

// ---------- stats pass 2: fold partials -> a = g*rsqrt(var+eps), bs = b - mean*a ----------
__global__ void bn_finalize(const float* __restrict__ psum, const float* __restrict__ psumsq,
                            const float* __restrict__ g, const float* __restrict__ b,
                            float* __restrict__ avec, float* __restrict__ bsvec,
                            int D, int nparts, float invM) {
  for (int d = threadIdx.x; d < D; d += blockDim.x) {
    float s = 0.f, s2 = 0.f;
    for (int i = 0; i < nparts; ++i) { s += psum[i * D + d]; s2 += psumsq[i * D + d]; }
    float mean = s * invM;
    float var  = s2 * invM - mean * mean;
    var = var > 0.f ? var : 0.f;
    float a = g[d] * rsqrtf(var + BN_EPS);
    avec[d]  = a;
    bsvec[d] = b[d] - mean * a;
  }
}

// ---------- fold BN into weights: Bt[n,k] = W[n,k]*a[k] (bf16), cp[n] = c[n] + sum_k W[n,k]*bs[k] ----
__global__ __launch_bounds__(256)
void prep_weights(const float* __restrict__ W, const float* __restrict__ c,
                  const float* __restrict__ avec, const float* __restrict__ bsvec,
                  __bf16* __restrict__ Bt, float* __restrict__ cp, int K) {
  const int n = blockIdx.x;
  const int tid = threadIdx.x;
  float dot = 0.f;
  for (int k = tid; k < K; k += 256) {
    float w = W[(size_t)n * K + k];
    Bt[(size_t)n * K + k] = (__bf16)(w * avec[k]);
    dot += w * bsvec[k];
  }
  __shared__ float red[256];
  red[tid] = dot; __syncthreads();
  for (int s = 128; s > 0; s >>= 1) {
    if (tid < s) red[tid] += red[tid + s];
    __syncthreads();
  }
  if (tid == 0) cp[n] = c[n] + red[0];
}

__global__ void f32_to_bf16(const float* __restrict__ in, __bf16* __restrict__ out, size_t n) {
  size_t i = (size_t)blockIdx.x * blockDim.x + threadIdx.x;
  size_t stride = (size_t)gridDim.x * blockDim.x;
  for (; i < n; i += stride) out[i] = (__bf16)in[i];
}

__global__ void init_neg_inf(unsigned int* out, int n) {
  int i = blockIdx.x * blockDim.x + threadIdx.x;
  if (i < n) out[i] = 0xFF800000u;   // -inf
}

// ---------- launch ----------
extern "C" void kernel_launch(void* const* d_in, const int* in_sizes, int n_in,
                              void* d_out, int out_size, void* d_ws, size_t ws_size,
                              hipStream_t stream) {
  (void)in_sizes; (void)n_in; (void)out_size; (void)ws_size;
  const int Bb = 32, Nset = 2048, M = Bb * Nset;    // 65536 rows
  const int D0 = 256, D1 = 512, D2 = 512, D3 = 1024;

  const float* x_in = (const float*)d_in[0];
  const float* g0 = (const float*)d_in[1];
  const float* b0 = (const float*)d_in[2];
  const float* W0 = (const float*)d_in[3];
  const float* c0 = (const float*)d_in[4];
  const float* g1 = (const float*)d_in[5];
  const float* b1 = (const float*)d_in[6];
  const float* W1 = (const float*)d_in[7];
  const float* c1 = (const float*)d_in[8];
  const float* g2 = (const float*)d_in[9];
  const float* b2 = (const float*)d_in[10];
  const float* W2 = (const float*)d_in[11];
  const float* c2 = (const float*)d_in[12];
  float* out = (float*)d_out;

  char* ws = (char*)d_ws;
  size_t off = 0;
  auto carve = [&](size_t bytes) { char* p = ws + off; off += (bytes + 255) & ~(size_t)255; return p; };
  __bf16* x0   = (__bf16*)carve((size_t)M * D0 * 2);   // 32 MB
  __bf16* act1 = (__bf16*)carve((size_t)M * D1 * 2);   // 64 MB
  __bf16* act2 = (__bf16*)carve((size_t)M * D2 * 2);   // 64 MB
  __bf16* Btw  = (__bf16*)carve((size_t)D3 * D2 * 2);  // 1 MB (largest layer)
  float*  cp    = (float*)carve((size_t)D3 * 4);
  const int NPARTS = 32;
  float*  psum  = (float*)carve((size_t)NPARTS * 512 * 4);
  float*  psq   = (float*)carve((size_t)NPARTS * 512 * 4);
  float*  avec  = (float*)carve(512 * 4);
  float*  bsvec = (float*)carve(512 * 4);

  const float invM = 1.0f / (float)M;
  const int RPB = M / NPARTS;   // 2048 rows per stats block

  init_neg_inf<<<(Bb * D3 + 255) / 256, 256, 0, stream>>>((unsigned int*)out, Bb * D3);
  f32_to_bf16<<<2048, 256, 0, stream>>>(x_in, x0, (size_t)M * D0);

  // ---- layer 0: BN(x0) @ W0^T + c0, ReLU -> act1 ----
  colstats_part<<<dim3(D0 / 64, NPARTS), 256, 0, stream>>>(x0, psum, psq, D0, RPB);
  bn_finalize<<<1, 256, 0, stream>>>(psum, psq, g0, b0, avec, bsvec, D0, NPARTS, invM);
  prep_weights<<<D1, 256, 0, stream>>>(W0, c0, avec, bsvec, Btw, cp, D0);
  gemm_bf16<true><<<dim3(D1 / 128, M / 128), 256, 0, stream>>>(x0, Btw, cp, act1, nullptr, D0, D1);

  // ---- layer 1 ----
  colstats_part<<<dim3(D1 / 64, NPARTS), 256, 0, stream>>>(act1, psum, psq, D1, RPB);
  bn_finalize<<<1, 256, 0, stream>>>(psum, psq, g1, b1, avec, bsvec, D1, NPARTS, invM);
  prep_weights<<<D2, 256, 0, stream>>>(W1, c1, avec, bsvec, Btw, cp, D1);
  gemm_bf16<true><<<dim3(D2 / 128, M / 128), 256, 0, stream>>>(act1, Btw, cp, act2, nullptr, D1, D2);

  // ---- layer 2 + fused set-max (layer-3 activations never hit HBM) ----
  colstats_part<<<dim3(D2 / 64, NPARTS), 256, 0, stream>>>(act2, psum, psq, D2, RPB);
  bn_finalize<<<1, 256, 0, stream>>>(psum, psq, g2, b2, avec, bsvec, D2, NPARTS, invM);
  prep_weights<<<D3, 256, 0, stream>>>(W2, c2, avec, bsvec, Btw, cp, D2);
  gemm_bf16<false><<<dim3(D3 / 128, M / 128), 256, 0, stream>>>(act2, Btw, cp, nullptr, out, D2, D3);
}
